// MaskedLatentModel_45354854646377
// MI455X (gfx1250) — matrix operand order repaired and checked by reading, not compiled, not run on hardware
//
#include <hip/hip_runtime.h>
#include <hip/hip_bf16.h>
#include <cstdint>
#include <cstddef>

// ---------------------------------------------------------------------------
// MaskedLatentModel forward for MI455X (gfx1250, wave32, WMMA).
// All heavy GEMMs run through v_wmma_f32_16x16x32_bf16 (bf16 in, f32 acc).
// ---------------------------------------------------------------------------

typedef __bf16 bf16;
typedef __attribute__((ext_vector_type(16))) __bf16 v16bf;
typedef __attribute__((ext_vector_type(8)))  float  v8f;

#define BDIM   32
#define CCH    3
#define IMGS   256
#define PS     16
#define LTOK   256         // tokens
#define DIM    768
#define HEADS  12
#define DHEAD  64
#define DFFN   3072
#define DEPTH  12
#define INNER  768
#define CTX    768
#define TLEN   77
#define TLENP  96          // padded key length for cross attention (mult of 32)
#define LENKEEP 64
#define BL     (BDIM * LTOK)      // 8192 rows of x
#define BT     (BDIM * TLEN)      // 2464 rows of text
#define BTPAD  2560               // BT padded to GEMM block-row multiple

__device__ __forceinline__ int wmma_k0(int v, int half) {
  // 16-bit A-matrix 16x32 layout: VGPR v holds a K pair; lanes 16-31 offset 8.
  return ((v & 4) ? 16 : 0) + half * 8 + 2 * (v & 3);
}

__device__ __forceinline__ v8f wmma_bf16(v16bf a, v16bf b, v8f c) {
  return __builtin_amdgcn_wmma_f32_16x16x32_bf16(false, a, false, b, (short)0, c,
                                                 false, false);
}

// ---------------------------------------------------------------------------
// Generic GEMM: C[M,N] = A[M,K](bf16,row-major) * B[K,N](bf16,row-major)
// Block tile 128x64, K-tile 64. 8 waves, each owns a 32x32 macro-tile
// (2 A frags x 2 B frags -> 4 WMMAs per 32-wide K slice).
// A buffer rows must be allocated up to ceil(M/128)*128 (caller guarantees);
// out-of-M rows are computed on garbage but never stored.
// Epilogue specialized at compile time: (+bias) -> (gelu?) -> (+resid)
// -> store f32 / bf16 / bf16-transposed-per-head (for attention K operand).
// STT store: row = b*Lq + j, col = h*64+d  ->  KT[(b*INNER + col)*LkPadT + j]
// ---------------------------------------------------------------------------
#define GBM 128
#define GBN 64
#define GBK 64
#define GPAD 8   // bf16 elements of row padding (16 bytes) keeps b128 alignment

template <bool HAS_BIAS, bool GELU, bool HAS_RESID, bool STF, bool STH, bool STT>
__global__ __launch_bounds__(256) void gemm_kernel(
    const bf16* __restrict__ A, const bf16* __restrict__ B,
    int M, int N, int K,
    const float* __restrict__ bias,
    float* __restrict__ outf, bf16* __restrict__ outh,
    const float* __restrict__ resid, int LqT, int LkPadT) {
  __shared__ __align__(16) bf16 As[GBM][GBK + GPAD];
  __shared__ __align__(16) bf16 Bs[GBK][GBN + GPAD];

  const int tid  = threadIdx.x;
  const int wave = tid >> 5;
  const int lane = tid & 31;
  const int m16  = lane & 15;
  const int half = lane >> 4;
  const int rowBase = blockIdx.y * GBM;
  const int colBase = blockIdx.x * GBN;
  const int rbase = (wave >> 1) * 32;   // wave row offset inside block tile
  const int cbase = (wave & 1) * 32;    // wave col offset inside block tile

  v8f acc[2][2] = {{{}, {}}, {{}, {}}};

  for (int k0t = 0; k0t < K; k0t += GBK) {
    // stage A tile (128x64 bf16): 4 x b128 per thread
#pragma unroll
    for (int i = 0; i < 4; ++i) {
      int ch = tid + 256 * i;
      int arow = ch >> 3;           // 8 chunks of 8 bf16 per row
      int acol = (ch & 7) * 8;
      uint4 val = *(const uint4*)(A + (size_t)(rowBase + arow) * K + k0t + acol);
      *(uint4*)(&As[arow][acol]) = val;
    }
    // stage B tile (64x64 bf16): 2 x b128 per thread
#pragma unroll
    for (int i = 0; i < 2; ++i) {
      int ch = tid + 256 * i;
      int brow = ch >> 3;
      int bcol = (ch & 7) * 8;
      uint4 val = *(const uint4*)(B + (size_t)(k0t + brow) * N + colBase + bcol);
      *(uint4*)(&Bs[brow][bcol]) = val;
    }
    __syncthreads();

#pragma unroll
    for (int ks = 0; ks < GBK; ks += 32) {
      v16bf a[2], bf[2];
#pragma unroll
      for (int s = 0; s < 2; ++s) {
#pragma unroll
        for (int v = 0; v < 8; ++v) {
          int kk = ks + wmma_k0(v, half);
          a[s][2 * v]     = As[rbase + s * 16 + m16][kk];
          a[s][2 * v + 1] = As[rbase + s * 16 + m16][kk + 1];
        }
#pragma unroll
        for (int j = 0; j < 16; ++j)
          bf[s][j] = Bs[ks + wmma_k0(j >> 1, half) + (j & 1)][cbase + s * 16 + m16];
      }
      acc[0][0] = wmma_bf16(a[0], bf[0], acc[0][0]);
      acc[0][1] = wmma_bf16(a[0], bf[1], acc[0][1]);
      acc[1][0] = wmma_bf16(a[1], bf[0], acc[1][0]);
      acc[1][1] = wmma_bf16(a[1], bf[1], acc[1][1]);
    }
    __syncthreads();
  }

#pragma unroll
  for (int sr = 0; sr < 2; ++sr) {
#pragma unroll
    for (int v = 0; v < 8; ++v) {
      int grow = rowBase + rbase + sr * 16 + v + 8 * half;
      if (grow >= M) continue;
#pragma unroll
      for (int sc = 0; sc < 2; ++sc) {
        int gcol = colBase + cbase + sc * 16 + m16;
        float val = acc[sr][sc][v];
        if (HAS_BIAS) val += bias[gcol];
        if (GELU) val = 0.5f * val * (1.0f + erff(val * 0.7071067811865475f));
        size_t oi = (size_t)grow * N + gcol;
        if (HAS_RESID) val += resid[oi];
        if (STF) outf[oi] = val;
        if (STH) outh[oi] = (bf16)val;
        if (STT) {
          int bb = grow / LqT;
          int jj = grow - bb * LqT;
          outh[((size_t)bb * INNER + gcol) * LkPadT + jj] = (bf16)val;
        }
      }
    }
  }
}

// ---------------------------------------------------------------------------
// Attention scores: S[bh,i,j] = (Q_bh[i,:] . K_bh[j,:]) / sqrt(dhead).
// KT layout: KT[(b*INNER + h*64 + d)*LkPad + j]  (contiguous in j).
// Loads are unguarded: pad columns hold finite garbage, overwritten by
// -FLT_MAX at the store stage (and by the text mask for cross attention).
// ---------------------------------------------------------------------------
__global__ __launch_bounds__(32) void attn_score_kernel(
    const bf16* __restrict__ Q, const bf16* __restrict__ KT,
    float* __restrict__ S, const unsigned char* __restrict__ mask,
    int Lk, int LkPad, float scale_inv) {
  const int bh = blockIdx.x;
  const int b = bh / HEADS, h = bh % HEADS;
  const int i0 = blockIdx.y * 16;
  const int j0 = blockIdx.z * 16;
  const int lane = threadIdx.x;
  const int m16 = lane & 15, half = lane >> 4;

  v8f acc = {};
  const bf16* qrow = Q + ((size_t)(b * LTOK + i0 + m16)) * INNER + h * DHEAD;
  const bf16* ktbase = KT + ((size_t)b * INNER + h * DHEAD) * LkPad + j0;

#pragma unroll
  for (int kk = 0; kk < DHEAD; kk += 32) {
    v16bf a, bfrag;
#pragma unroll
    for (int v = 0; v < 8; ++v) {
      int k = kk + wmma_k0(v, half);
      a[2 * v]     = qrow[k];
      a[2 * v + 1] = qrow[k + 1];
    }
#pragma unroll
    for (int j = 0; j < 16; ++j)
      bfrag[j] = ktbase[(size_t)(kk + wmma_k0(j >> 1, half) + (j & 1)) * LkPad + m16];
    acc = wmma_bf16(a, bfrag, acc);
  }

  const int jcol = j0 + m16;
  const bool valid = (jcol < Lk) && (!mask || mask[b * TLEN + jcol] != 0);
#pragma unroll
  for (int v = 0; v < 8; ++v) {
    int irow = i0 + v + 8 * half;
    float val = valid ? acc[v] * scale_inv : -3.402823466e38f;
    S[((size_t)bh * LTOK + irow) * LkPad + jcol] = val;
  }
}

// O[b, i, h*64+d] = sum_j P[bh,i,j] * V[b,j,h*64+d]
// Unguarded: pad columns of P are exact zeros; V rows stay in-bounds.
__global__ __launch_bounds__(32) void attn_av_kernel(
    const bf16* __restrict__ P, const bf16* __restrict__ V,
    bf16* __restrict__ O, int Lk, int LkPad) {
  const int bh = blockIdx.x;
  const int b = bh / HEADS, h = bh % HEADS;
  const int i0 = blockIdx.y * 16;
  const int d0 = blockIdx.z * 16;
  const int lane = threadIdx.x;
  const int m16 = lane & 15, half = lane >> 4;

  v8f acc = {};
  const bf16* prow = P + ((size_t)bh * LTOK + i0 + m16) * LkPad;

  for (int kk = 0; kk < LkPad; kk += 32) {
    v16bf a, bfrag;
#pragma unroll
    for (int v = 0; v < 8; ++v) {
      int k = kk + wmma_k0(v, half);
      a[2 * v]     = prow[k];
      a[2 * v + 1] = prow[k + 1];
    }
#pragma unroll
    for (int jj = 0; jj < 16; ++jj) {
      int j = kk + wmma_k0(jj >> 1, half) + (jj & 1);
      bfrag[jj] = V[((size_t)(b * Lk + j)) * INNER + h * DHEAD + d0 + m16];
    }
    acc = wmma_bf16(a, bfrag, acc);
  }

#pragma unroll
  for (int v = 0; v < 8; ++v) {
    int irow = i0 + v + 8 * half;
    O[((size_t)(b * LTOK + irow)) * INNER + h * DHEAD + d0 + m16] = (bf16)acc[v];
  }
}

// row softmax over width LkPad (<=256); pad/masked entries carry -FLT_MAX
__global__ __launch_bounds__(256) void softmax_kernel(
    const float* __restrict__ S, bf16* __restrict__ P, int LkPad) {
  __shared__ float red[256];
  const size_t row = blockIdx.x;
  const float* srow = S + row * LkPad;
  bf16* prow = P + row * LkPad;
  const int t = threadIdx.x;

  float m = -3.402823466e38f;
  if (t < LkPad) m = srow[t];
  red[t] = m;
  __syncthreads();
  for (int s = 128; s > 0; s >>= 1) {
    if (t < s) red[t] = fmaxf(red[t], red[t + s]);
    __syncthreads();
  }
  m = red[0];
  __syncthreads();

  float e = (t < LkPad) ? __expf(srow[t] - m) : 0.0f;
  red[t] = e;
  __syncthreads();
  for (int s = 128; s > 0; s >>= 1) {
    if (t < s) red[t] += red[t + s];
    __syncthreads();
  }
  float inv = 1.0f / red[0];
  if (t < LkPad) prow[t] = (bf16)(e * inv);
}

// ---------------------------------------------------------------------------
// LayerNorm (f32 in -> bf16 out), one row per block
// ---------------------------------------------------------------------------
__global__ __launch_bounds__(256) void ln_kernel(
    const float* __restrict__ X, const float* __restrict__ w,
    const float* __restrict__ b, bf16* __restrict__ out, int cols) {
  __shared__ float red[256];
  const size_t row = blockIdx.x;
  const float* xr = X + row * cols;
  const int t = threadIdx.x;

  float s = 0.f, ss = 0.f;
  for (int j = t; j < cols; j += 256) {
    float v = xr[j];
    s += v;
    ss += v * v;
  }
  red[t] = s;
  __syncthreads();
  for (int k = 128; k > 0; k >>= 1) {
    if (t < k) red[t] += red[t + k];
    __syncthreads();
  }
  float mu = red[0] / (float)cols;
  __syncthreads();
  red[t] = ss;
  __syncthreads();
  for (int k = 128; k > 0; k >>= 1) {
    if (t < k) red[t] += red[t + k];
    __syncthreads();
  }
  float var = red[0] / (float)cols - mu * mu;
  float rs = rsqrtf(var + 1e-5f);
  for (int j = t; j < cols; j += 256) {
    out[row * cols + j] = (bf16)((xr[j] - mu) * rs * w[j] + b[j]);
  }
}

// ---------------------------------------------------------------------------
// Masking: rank_i = #{j : n_j < n_i or (n_j == n_i and j < i)}; keep = rank<64
// ---------------------------------------------------------------------------
__global__ __launch_bounds__(256) void keep_kernel(const float* __restrict__ noise,
                                                   int* __restrict__ keep) {
  __shared__ float nz[LTOK];
  const int b = blockIdx.x, i = threadIdx.x;
  nz[i] = noise[b * LTOK + i];
  __syncthreads();
  float ni = nz[i];
  int rank = 0;
  for (int j = 0; j < LTOK; ++j) {
    float nj = nz[j];
    rank += (nj < ni) || (nj == ni && j < i);
  }
  keep[b * LTOK + i] = (rank < LENKEEP) ? 1 : 0;
}

__global__ __launch_bounds__(256) void assemble_kernel(
    const float* __restrict__ pe, const float* __restrict__ convb,
    const float* __restrict__ mtok, const float* __restrict__ pos,
    const int* __restrict__ keep, float* __restrict__ X) {
  const size_t row = blockIdx.x;  // b*256 + l
  const int l = (int)(row & 255);
  const int k = keep[row];
  for (int d = threadIdx.x; d < DIM; d += 256) {
    float v = k ? (pe[row * DIM + d] + convb[d]) : mtok[d];
    X[row * DIM + d] = v + pos[(size_t)l * DIM + d];
  }
}

// patches (b,l,(c,p,q)) <- img, as bf16 GEMM operand
__global__ __launch_bounds__(256) void patches_kernel(const float* __restrict__ img,
                                                      bf16* __restrict__ Pm) {
  const size_t row = blockIdx.x;  // b*256 + hh*16 + ww
  const int b = (int)(row >> 8);
  const int l = (int)(row & 255);
  const int hh = l >> 4, ww = l & 15;
  for (int idx = threadIdx.x; idx < DIM; idx += 256) {
    int c = idx >> 8, r = idx & 255, p = r >> 4, q = r & 15;
    Pm[row * DIM + idx] =
        (bf16)img[(((size_t)b * CCH + c) * IMGS + hh * PS + p) * IMGS + ww * PS + q];
  }
}

__global__ __launch_bounds__(256) void cvt_kernel(const float* __restrict__ in,
                                                  bf16* __restrict__ out, size_t n) {
  size_t i = (size_t)blockIdx.x * 256 + threadIdx.x;
  if (i < n) out[i] = (bf16)in[i];
}

// conv_w [D][K] f32 -> Wt [K][D] bf16
__global__ __launch_bounds__(256) void tcvt_kernel(const float* __restrict__ in,
                                                   bf16* __restrict__ out, int D,
                                                   int K) {
  size_t i = (size_t)blockIdx.x * 256 + threadIdx.x;
  if (i >= (size_t)D * K) return;
  int d = (int)(i / K), k = (int)(i % K);
  out[(size_t)k * D + d] = (bf16)in[i];
}

__global__ void zero_kernel(float* p) { p[0] = 0.0f; }

// loss = mean((xrec - patchify(img))^2), patch col order = (p, q, c)
__global__ __launch_bounds__(256) void loss_kernel(const float* __restrict__ xrec,
                                                   const float* __restrict__ img,
                                                   float* __restrict__ loss) {
  __shared__ float red[256];
  const size_t total = (size_t)BL * DIM;
  size_t idx = (size_t)blockIdx.x * 256 + threadIdx.x;
  float d2 = 0.f;
  if (idx < total) {
    size_t row = idx / DIM;
    int col = (int)(idx % DIM);
    int b = (int)(row >> 8);
    int l = (int)(row & 255);
    int hh = l >> 4, ww = l & 15;
    int p = col / 48, rem = col % 48, q = rem / 3, c = rem % 3;
    float tgt =
        img[(((size_t)b * CCH + c) * IMGS + hh * PS + p) * IMGS + ww * PS + q];
    float diff = xrec[idx] - tgt;
    d2 = diff * diff;
  }
  red[threadIdx.x] = d2;
  __syncthreads();
  for (int s = 128; s > 0; s >>= 1) {
    if (threadIdx.x < s) red[threadIdx.x] += red[threadIdx.x + s];
    __syncthreads();
  }
  if (threadIdx.x == 0)
    atomicAdd(loss, red[0] * (1.0f / ((float)BL * (float)DIM)));
}

// ---------------------------------------------------------------------------
extern "C" void kernel_launch(void* const* d_in, const int* in_sizes, int n_in,
                              void* d_out, int out_size, void* d_ws, size_t ws_size,
                              hipStream_t stream) {
  (void)in_sizes; (void)n_in; (void)out_size; (void)ws_size;

  const float* img   = (const float*)d_in[0];
  const float* text  = (const float*)d_in[1];
  const unsigned char* tmask = (const unsigned char*)d_in[2];  // bool array
  const float* noise = (const float*)d_in[3];
  const float* conv_w = (const float*)d_in[4];
  const float* conv_b = (const float*)d_in[5];
  const float* mtok   = (const float*)d_in[6];
  const float* posi   = (const float*)d_in[7];
  const float* ln1w = (const float*)d_in[8];
  const float* ln1b = (const float*)d_in[9];
  const float* wq1  = (const float*)d_in[10];
  const float* wk1  = (const float*)d_in[11];
  const float* wv1  = (const float*)d_in[12];
  const float* wo1  = (const float*)d_in[13];
  const float* ln2w = (const float*)d_in[14];
  const float* ln2b = (const float*)d_in[15];
  const float* ln3w = (const float*)d_in[16];
  const float* ln3b = (const float*)d_in[17];
  const float* wq2  = (const float*)d_in[18];
  const float* wk2  = (const float*)d_in[19];
  const float* wv2  = (const float*)d_in[20];
  const float* wo2  = (const float*)d_in[21];
  const float* ln4w = (const float*)d_in[22];
  const float* ln4b = (const float*)d_in[23];
  const float* w1   = (const float*)d_in[24];
  const float* b1   = (const float*)d_in[25];
  const float* w2   = (const float*)d_in[26];
  const float* b2   = (const float*)d_in[27];
  const float* decw = (const float*)d_in[28];
  const float* decb = (const float*)d_in[29];

  float* fout = (float*)d_out;        // fout[0] = loss, fout+1 = xrec
  float* xrec = fout + 1;

  // ---- workspace layout (~550 MB) ----
  char* ws = (char*)d_ws;
  size_t off = 0;
  auto alloc = [&](size_t bytes) -> void* {
    void* p = ws + off;
    off = (off + bytes + 255) & ~(size_t)255;
    return p;
  };
  const size_t WSQ = (size_t)DEPTH * DIM * INNER;     // 7,077,888
  const size_t WSF = (size_t)DEPTH * DIM * DFFN;      // 28,311,552
  bf16* wq1h = (bf16*)alloc(WSQ * 2);
  bf16* wk1h = (bf16*)alloc(WSQ * 2);
  bf16* wv1h = (bf16*)alloc(WSQ * 2);
  bf16* wo1h = (bf16*)alloc(WSQ * 2);
  bf16* wq2h = (bf16*)alloc(WSQ * 2);
  bf16* wk2h = (bf16*)alloc(WSQ * 2);
  bf16* wv2h = (bf16*)alloc(WSQ * 2);
  bf16* wo2h = (bf16*)alloc(WSQ * 2);
  bf16* w1h  = (bf16*)alloc(WSF * 2);
  bf16* w2h  = (bf16*)alloc(WSF * 2);
  bf16* convWt = (bf16*)alloc((size_t)DIM * DIM * 2);
  bf16* decWh  = (bf16*)alloc((size_t)DIM * DIM * 2);
  bf16* Pm  = (bf16*)alloc((size_t)BL * DIM * 2);
  float* PEf = (float*)alloc((size_t)BL * DIM * 4);
  int* keep = (int*)alloc((size_t)BL * 4);
  float* Xf = (float*)alloc((size_t)BL * DIM * 4);
  bf16* Hh  = (bf16*)alloc((size_t)BL * DIM * 2);
  bf16* Tln = (bf16*)alloc((size_t)BTPAD * CTX * 2);  // padded: GEMM reads 2560 rows
  bf16* Qh  = (bf16*)alloc((size_t)BL * INNER * 2);
  bf16* KTh = (bf16*)alloc((size_t)BDIM * INNER * LTOK * 2);  // transposed K
  bf16* Vh  = (bf16*)alloc((size_t)BL * INNER * 2);
  float* Sbuf = (float*)alloc((size_t)BDIM * HEADS * LTOK * LTOK * 4);
  bf16* Pbuf  = (bf16*)alloc((size_t)BDIM * HEADS * LTOK * LTOK * 2);
  bf16* Obuf  = (bf16*)alloc((size_t)BL * INNER * 2);
  bf16* Gbuf  = (bf16*)alloc((size_t)BL * DFFN * 2);

  auto cvt = [&](const float* in, bf16* out, size_t n) {
    cvt_kernel<<<(unsigned)((n + 255) / 256), 256, 0, stream>>>(in, out, n);
  };
  auto ggrid = [&](int M, int N) {
    return dim3((N + GBN - 1) / GBN, (M + GBM - 1) / GBM);
  };
  // specialized GEMM launchers (compile-time epilogues)
  auto gemm_f = [&](const bf16* A, const bf16* Bm, int M, int N, int K,
                    float* outf) {  // plain f32 store
    gemm_kernel<false, false, false, true, false, false>
        <<<ggrid(M, N), 256, 0, stream>>>(A, Bm, M, N, K, nullptr, outf, nullptr,
                                          nullptr, 0, 0);
  };
  auto gemm_h = [&](const bf16* A, const bf16* Bm, int M, int N, int K,
                    bf16* outh) {   // plain bf16 store (Q/V)
    gemm_kernel<false, false, false, false, true, false>
        <<<ggrid(M, N), 256, 0, stream>>>(A, Bm, M, N, K, nullptr, nullptr, outh,
                                          nullptr, 0, 0);
  };
  auto gemm_kT = [&](const bf16* A, const bf16* Bm, int M, int N, int K,
                     bf16* outT, int Lq, int LkPadT) {  // K proj, transposed
    gemm_kernel<false, false, false, false, false, true>
        <<<ggrid(M, N), 256, 0, stream>>>(A, Bm, M, N, K, nullptr, nullptr, outT,
                                          nullptr, Lq, LkPadT);
  };
  auto gemm_res = [&](const bf16* A, const bf16* Bm, int M, int N, int K,
                      float* xio) {  // x += A*B (output projections)
    gemm_kernel<false, false, true, true, false, false>
        <<<ggrid(M, N), 256, 0, stream>>>(A, Bm, M, N, K, nullptr, xio, nullptr,
                                          xio, 0, 0);
  };
  auto gemm_gelu = [&](const bf16* A, const bf16* Bm, int M, int N, int K,
                       const float* bias, bf16* outh) {  // FFN1
    gemm_kernel<true, true, false, false, true, false>
        <<<ggrid(M, N), 256, 0, stream>>>(A, Bm, M, N, K, bias, nullptr, outh,
                                          nullptr, 0, 0);
  };
  auto gemm_bres = [&](const bf16* A, const bf16* Bm, int M, int N, int K,
                       const float* bias, float* xio) {  // FFN2: x += A*B + b
    gemm_kernel<true, false, true, true, false, false>
        <<<ggrid(M, N), 256, 0, stream>>>(A, Bm, M, N, K, bias, xio, nullptr, xio,
                                          0, 0);
  };
  auto gemm_bias = [&](const bf16* A, const bf16* Bm, int M, int N, int K,
                       const float* bias, float* outf) {  // decoder
    gemm_kernel<true, false, false, true, false, false>
        <<<ggrid(M, N), 256, 0, stream>>>(A, Bm, M, N, K, bias, outf, nullptr,
                                          nullptr, 0, 0);
  };

  // ---- weight conversion (f32 -> bf16) ----
  cvt(wq1, wq1h, WSQ);  cvt(wk1, wk1h, WSQ);  cvt(wv1, wv1h, WSQ);
  cvt(wo1, wo1h, WSQ);  cvt(wq2, wq2h, WSQ);  cvt(wk2, wk2h, WSQ);
  cvt(wv2, wv2h, WSQ);  cvt(wo2, wo2h, WSQ);
  cvt(w1, w1h, WSF);    cvt(w2, w2h, WSF);
  cvt(decw, decWh, (size_t)DIM * DIM);
  tcvt_kernel<<<(DIM * DIM + 255) / 256, 256, 0, stream>>>(conv_w, convWt, DIM, DIM);

  // ---- patch embed ----
  patches_kernel<<<BL, 256, 0, stream>>>(img, Pm);
  gemm_f(Pm, convWt, BL, DIM, DIM, PEf);
  keep_kernel<<<BDIM, 256, 0, stream>>>(noise, keep);
  assemble_kernel<<<BL, 256, 0, stream>>>(PEf, conv_b, mtok, posi, keep, Xf);

  const float scale_inv = 0.125f;  // 1/sqrt(64)
  for (int l = 0; l < DEPTH; ++l) {
    const size_t oq = (size_t)l * DIM * INNER;
    const size_t of = (size_t)l * DIM * DFFN;
    // ---- self attention ----
    ln_kernel<<<BL, 256, 0, stream>>>(Xf, ln1w + l * DIM, ln1b + l * DIM, Hh, DIM);
    gemm_h(Hh, wq1h + oq, BL, INNER, DIM, Qh);
    gemm_kT(Hh, wk1h + oq, BL, INNER, DIM, KTh, LTOK, LTOK);
    gemm_h(Hh, wv1h + oq, BL, INNER, DIM, Vh);
    attn_score_kernel<<<dim3(BDIM * HEADS, LTOK / 16, LTOK / 16), 32, 0, stream>>>(
        Qh, KTh, Sbuf, nullptr, LTOK, LTOK, scale_inv);
    softmax_kernel<<<BDIM * HEADS * LTOK, 256, 0, stream>>>(Sbuf, Pbuf, LTOK);
    attn_av_kernel<<<dim3(BDIM * HEADS, LTOK / 16, DHEAD / 16), 32, 0, stream>>>(
        Pbuf, Vh, Obuf, LTOK, LTOK);
    gemm_res(Obuf, wo1h + oq, BL, DIM, INNER, Xf);
    // ---- cross attention ----
    ln_kernel<<<BL, 256, 0, stream>>>(Xf, ln2w + l * DIM, ln2b + l * DIM, Hh, DIM);
    ln_kernel<<<BT, 256, 0, stream>>>(text, ln3w + l * CTX, ln3b + l * CTX, Tln, CTX);
    gemm_h(Hh, wq2h + oq, BL, INNER, DIM, Qh);
    gemm_kT(Tln, wk2h + oq, BT, INNER, CTX, KTh, TLEN, TLENP);
    gemm_h(Tln, wv2h + oq, BT, INNER, CTX, Vh);
    attn_score_kernel<<<dim3(BDIM * HEADS, LTOK / 16, TLENP / 16), 32, 0, stream>>>(
        Qh, KTh, Sbuf, tmask, TLEN, TLENP, scale_inv);
    softmax_kernel<<<BDIM * HEADS * LTOK, 256, 0, stream>>>(Sbuf, Pbuf, TLENP);
    attn_av_kernel<<<dim3(BDIM * HEADS, LTOK / 16, DHEAD / 16), 32, 0, stream>>>(
        Pbuf, Vh, Obuf, TLEN, TLENP);
    gemm_res(Obuf, wo2h + oq, BL, DIM, INNER, Xf);
    // ---- FFN ----
    ln_kernel<<<BL, 256, 0, stream>>>(Xf, ln4w + l * DIM, ln4b + l * DIM, Hh, DIM);
    gemm_gelu(Hh, w1h + of, BL, DFFN, DIM, b1 + l * DFFN, Gbuf);
    gemm_bres(Gbuf, w2h + of, BL, DIM, DFFN, b2 + l * DIM, Xf);
  }

  // ---- decoder + loss ----
  cvt(Xf, Hh, (size_t)BL * DIM);
  gemm_bias(Hh, decWh, BL, DIM, DIM, decb, xrec);
  zero_kernel<<<1, 1, 0, stream>>>(fout);
  loss_kernel<<<(unsigned)(((size_t)BL * DIM + 255) / 256), 256, 0, stream>>>(
      xrec, img, fout);
}
